// ACMIL_MHA_35364760715273
// MI455X (gfx1250) — compile-verified, matrix operand order checked by reading, not run on hardware
//
#include <hip/hip_runtime.h>
#include <hip/hip_bf16.h>
#include <math.h>
#include <stdint.h>

// ---------------------------------------------------------------------------
// ACMIL MHA forward, algebraically reduced:
//   h    = relu(x @ W_dr + b_dr)                      (WMMA GEMM, 20000x1024x512)
//   attn = h @ Aw + c,  Aw = (Wk @ q)/8 per (t,head)  (WMMA GEMM, 20000x512x40)
//   topk+mask, softmax stats
//   g[t,h,:] = sum_n pa * h[n,:]   (streaming pass; V/K never materialized)
//   out/bag branches = tiny matvecs + layernorms
// GEMMs: double-buffered LDS, f16 operands staged with GLOBAL_LOAD_ASYNC_TO_LDS,
// f32 A operand converted in-flight (VALU) since conversion must happen anyway.
// ---------------------------------------------------------------------------

typedef __attribute__((ext_vector_type(16))) _Float16 v16h;
typedef __attribute__((ext_vector_type(8)))  _Float16 v8h;
typedef __attribute__((ext_vector_type(8)))  float    v8f;

#define NPTS   20000
#define DFEAT  1024
#define DIN    512
#define TOK    5
#define HEADS  8
#define NCLSS  2
#define NTH    (TOK*HEADS)   // 40
#define NTHP   64            // padded col count for attn GEMM
#define NCHUNK 40            // n-chunks for weighted sum (20000/500)
#define CHROWS 500

// async global -> LDS copy of 16 bytes per active lane (ASYNCcnt-tracked)
#define ASYNC_COPY_B128(ldsoff, gaddr)                                          \
    asm volatile("global_load_async_to_lds_b128 %0, %1, off"                    \
                 :: "v"(ldsoff), "v"(gaddr) : "memory")
#define WAIT_ASYNC0() asm volatile("s_wait_asynccnt 0x0" ::: "memory")

// ---------------------------------------------------------------------------
// Generic WMMA GEMM:  C[M,Ntot] = act(A[M,K] @ Bt^T + bias)
//   A:  f32 (converted to f16 during LDS staging) or f16 (async-staged)
//   Bt: f16, [Ntot][K] (K contiguous), async-staged
//   Output: row-major f16, or transposed f32 [validCols][M]
// ---------------------------------------------------------------------------
template<int BM, int BN, int WAVES_M, int WAVES_N, int A_IS_F32, int RELU, int STORE_TRANS>
__global__ __launch_bounds__(WAVES_M*WAVES_N*32)
void wmma_gemm(const void* __restrict__ Aptr, const _Float16* __restrict__ Bt,
               const float* __restrict__ bias, _Float16* __restrict__ Cf16,
               float* __restrict__ Ct, int M, int K, int Ntot, int validCols)
{
    constexpr int NTHREADS = WAVES_M*WAVES_N*32;
    constexpr int WM = BM / (WAVES_M*16);
    constexpr int WN = BN / (WAVES_N*16);

    __shared__ alignas(16) _Float16 As[2][BM*32];
    __shared__ alignas(16) _Float16 Bs[2][BN*32];

    const int tid  = threadIdx.x;
    const int row0 = blockIdx.x * BM;
    const int col0 = blockIdx.y * BN;
    const int wave = tid >> 5;
    const int lane = tid & 31;
    const int l16  = lane & 15;
    const int hl   = lane >> 4;
    const int wm   = wave / WAVES_N;
    const int wn   = wave % WAVES_N;

    v8f acc[WM][WN];
    const v8f vzero = {0.f,0.f,0.f,0.f,0.f,0.f,0.f,0.f};
    #pragma unroll
    for (int a = 0; a < WM; a++)
        #pragma unroll
        for (int b = 0; b < WN; b++) acc[a][b] = vzero;

    // ---- stage A tile (BM x 32 halves) into buffer `buf` for k-offset kt ----
    auto stageA = [&](int buf, int kt) {
        for (int s = tid; s < BM*2; s += NTHREADS) {
            const int row  = s >> 1;
            const int kseg = (s & 1) << 4;
            const int grow = row0 + row;
            if (A_IS_F32) {
                v8h t0, t1;
                if (grow < M) {
                    const float* src = (const float*)Aptr + (size_t)grow*K + kt + kseg;
                    if (kt + 32 < K) __builtin_prefetch((const void*)(src + 32), 0, 0);
                    #pragma unroll
                    for (int u = 0; u < 8; u++) {
                        t0[u] = (_Float16)src[u];
                        t1[u] = (_Float16)src[8+u];
                    }
                } else {
                    #pragma unroll
                    for (int u = 0; u < 8; u++) { t0[u] = (_Float16)0.f; t1[u] = (_Float16)0.f; }
                }
                *(v8h*)&As[buf][row*32 + kseg]     = t0;
                *(v8h*)&As[buf][row*32 + kseg + 8] = t1;
            } else {
                const int rcl = grow < M ? grow : 0;     // clamp; results discarded by store guard
                const _Float16* src = (const _Float16*)Aptr + (size_t)rcl*K + kt + kseg;
                const uint32_t ldsoff = (uint32_t)(uintptr_t)&As[buf][row*32 + kseg];
                const uint64_t ga = (uint64_t)(uintptr_t)src;
                ASYNC_COPY_B128(ldsoff, ga);
            }
        }
    };
    // ---- stage B tile (BN cols x 32 halves, K contiguous) via async LDS DMA ----
    auto stageB = [&](int buf, int kt) {
        for (int s = tid; s < BN*2; s += NTHREADS) {
            const int col  = s >> 1;
            const int kseg = (s & 1) << 4;
            const _Float16* src = Bt + (size_t)(col0 + col)*K + kt + kseg;
            const uint32_t ldsoff = (uint32_t)(uintptr_t)&Bs[buf][col*32 + kseg];
            const uint64_t ga = (uint64_t)(uintptr_t)src;
            ASYNC_COPY_B128(ldsoff, ga);
        }
    };

    stageA(0, 0);
    stageB(0, 0);
    WAIT_ASYNC0();
    __syncthreads();

    int buf = 0;
    for (int kt = 0; kt < K; kt += 32) {
        const int nkt = kt + 32;
        if (nkt < K) { stageA(buf ^ 1, nkt); stageB(buf ^ 1, nkt); }

        // ---- fragment loads per ISA 16-bit A/B layout ----
        v16h afrag[WM], bfrag[WN];
        #pragma unroll
        for (int fm = 0; fm < WM; fm++) {
            const int row = (wm*WM + fm)*16 + l16;
            const v8h lo = *(const v8h*)&As[buf][row*32 + hl*8];
            const v8h hi = *(const v8h*)&As[buf][row*32 + 16 + hl*8];
            #pragma unroll
            for (int u = 0; u < 8; u++) { afrag[fm][u] = lo[u]; afrag[fm][8+u] = hi[u]; }
        }
        #pragma unroll
        for (int fn = 0; fn < WN; fn++) {
            const int col = (wn*WN + fn)*16 + l16;
            const v8h lo = *(const v8h*)&Bs[buf][col*32 + hl*8];
            const v8h hi = *(const v8h*)&Bs[buf][col*32 + 16 + hl*8];
            #pragma unroll
            for (int u = 0; u < 8; u++) { bfrag[fn][u] = lo[u]; bfrag[fn][8+u] = hi[u]; }
        }
        #pragma unroll
        for (int fm = 0; fm < WM; fm++)
            #pragma unroll
            for (int fn = 0; fn < WN; fn++)
                acc[fm][fn] = __builtin_amdgcn_wmma_f32_16x16x32_f16(
                    false, afrag[fm], false, bfrag[fn], (short)0, acc[fm][fn], false, false);

        WAIT_ASYNC0();          // next-buffer async copies have landed in LDS
        __syncthreads();
        buf ^= 1;
    }

    // ---- epilogue: bias + relu, store ----
    #pragma unroll
    for (int fm = 0; fm < WM; fm++) {
        #pragma unroll
        for (int fn = 0; fn < WN; fn++) {
            const int rbase = row0 + (wm*WM + fm)*16 + hl*8;
            const int c     = col0 + (wn*WN + fn)*16 + l16;
            const float bv  = bias ? bias[c] : 0.f;
            #pragma unroll
            for (int r = 0; r < 8; r++) {
                const int gr = rbase + r;
                if (gr < M) {
                    float v = acc[fm][fn][r] + bv;
                    if (RELU) v = fmaxf(v, 0.f);
                    if (STORE_TRANS) {
                        if (c < validCols) Ct[(size_t)c*M + gr] = v;
                    } else {
                        Cf16[(size_t)gr*(size_t)Ntot + c] = (_Float16)v;
                    }
                }
            }
        }
    }
}

// ---------------------------------------------------------------------------
// Small helper kernels
// ---------------------------------------------------------------------------
__global__ void k_transpose_wdr(const float* __restrict__ Wdr, _Float16* __restrict__ WdrT) {
    int id = blockIdx.x*blockDim.x + threadIdx.x;   // id = j*1024 + i
    if (id >= DIN*DFEAT) return;
    int j = id >> 10, i = id & 1023;
    WdrT[id] = (_Float16)Wdr[(size_t)i*DIN + j];
}

__global__ void k_qvec(const float* __restrict__ q, const float* __restrict__ Wq,
                       const float* __restrict__ bq, float* __restrict__ qv) {
    int id = blockIdx.x*blockDim.x + threadIdx.x;
    if (id >= TOK*DIN) return;
    int t = id >> 9, j = id & 511;
    const float* qt = q + t*DIN;
    const float* W  = Wq + (size_t)t*DIN*DIN;
    float s = bq[t*DIN + j];
    for (int i = 0; i < DIN; i++) s = fmaf(qt[i], W[(size_t)i*DIN + j], s);
    qv[id] = s;
}

__global__ void k_awt(const float* __restrict__ Wk, const float* __restrict__ bk,
                      const float* __restrict__ qv, _Float16* __restrict__ AwT,
                      float* __restrict__ cA) {
    int id = blockIdx.x*blockDim.x + threadIdx.x;
    if (id >= NTHP*DIN) return;
    int th = id >> 9, i = id & 511;
    if (th >= NTH) { AwT[id] = (_Float16)0.f; if (i == 0) cA[th] = 0.f; return; }
    int t = th / HEADS, h = th % HEADS;
    const float* W  = Wk + (size_t)t*DIN*DIN + (size_t)i*DIN + h*64;
    const float* qh = qv + t*DIN + h*64;
    float s = 0.f;
    for (int d = 0; d < 64; d++) s = fmaf(W[d], qh[d], s);
    AwT[th*DIN + i] = (_Float16)(s * 0.125f);   // 1/sqrt(64)
    if (i == 0) {
        const float* bkh = bk + t*DIN + h*64;
        float c = 0.f;
        for (int d = 0; d < 64; d++) c = fmaf(bkh[d], qh[d], c);
        cA[th] = c * 0.125f;
    }
}

__device__ __forceinline__ unsigned hash_u32(unsigned x) {
    x ^= x >> 16; x *= 0x7feb352dU; x ^= x >> 15; x *= 0x846ca68bU; x ^= x >> 16;
    return x;
}

// one block per (t,head): deterministic top-10, then mask 5 of them (hash-ranked)
__global__ void k_topk_mask(float* __restrict__ attnW) {
    int th = blockIdx.x, tid = threadIdx.x;
    __shared__ float sv[256];
    __shared__ int   si[256];
    __shared__ int   found[10];
    float* row = attnW + (size_t)th*NPTS;
    for (int pass = 0; pass < 10; pass++) {
        float bv = -3.4e38f; int bi = 0;
        for (int n = tid; n < NPTS; n += 256) {
            bool ex = false;
            for (int j = 0; j < pass; j++) if (found[j] == n) ex = true;
            float v = row[n];
            if (!ex && v > bv) { bv = v; bi = n; }
        }
        sv[tid] = bv; si[tid] = bi; __syncthreads();
        for (int s = 128; s > 0; s >>= 1) {
            if (tid < s && sv[tid+s] > sv[tid]) { sv[tid] = sv[tid+s]; si[tid] = si[tid+s]; }
            __syncthreads();
        }
        if (tid == 0) found[pass] = si[0];
        __syncthreads();
    }
    if (tid == 0) {
        unsigned key[10]; bool used[10];
        for (int j = 0; j < 10; j++) { key[j] = hash_u32(th*131u + j*2654435761u); used[j] = false; }
        for (int s = 0; s < 5; s++) {
            int best = 0; unsigned bk = 0xffffffffu;
            for (int j = 0; j < 10; j++) if (!used[j] && key[j] < bk) { bk = key[j]; best = j; }
            used[best] = true;
            row[found[best]] = -1e9f;
        }
    }
}

// one block per (t,head): copy masked attn to d_out[H,T,N], compute softmax stats
__global__ void k_softmax_stats(const float* __restrict__ attnW, float* __restrict__ stats,
                                float* __restrict__ out_attn) {
    int th = blockIdx.x, t = th / HEADS, h = th % HEADS, tid = threadIdx.x;
    __shared__ float red[256];
    __shared__ float sh0;
    const float* row = attnW + (size_t)th*NPTS;
    float* dst = out_attn + (size_t)(h*TOK + t)*NPTS;
    float m = -3.4e38f;
    for (int n = tid; n < NPTS; n += 256) { float v = row[n]; dst[n] = v; m = fmaxf(m, v); }
    red[tid] = m; __syncthreads();
    for (int s = 128; s > 0; s >>= 1) { if (tid < s) red[tid] = fmaxf(red[tid], red[tid+s]); __syncthreads(); }
    if (tid == 0) sh0 = red[0];
    __syncthreads();
    m = sh0;
    float sum = 0.f;
    for (int n = tid; n < NPTS; n += 256) sum += __expf(row[n] - m);
    red[tid] = sum; __syncthreads();
    for (int s = 128; s > 0; s >>= 1) { if (tid < s) red[tid] += red[tid+s]; __syncthreads(); }
    if (tid == 0) { stats[th*2] = m; stats[th*2+1] = 1.f / red[0]; }
}

// streaming weighted sums: g_part[chunk][th][i] = sum_{n in chunk} pa[th,n]*h[n,i]
__global__ void k_wsum(const _Float16* __restrict__ hf, const float* __restrict__ attnW,
                       const float* __restrict__ stats, float* __restrict__ g_part) {
    const int chunk = blockIdx.x;
    const int ic    = blockIdx.y;
    const int tid   = threadIdx.x;
    __shared__ float spa[NTH*64];
    const int i  = ic*256 + tid;
    const int n0 = chunk*CHROWS;
    float acc[NTH];
    #pragma unroll
    for (int th = 0; th < NTH; th++) acc[th] = 0.f;

    for (int tt = 0; tt < CHROWS; tt += 64) {
        const int cnt = (CHROWS - tt) < 64 ? (CHROWS - tt) : 64;
        for (int e = tid; e < NTH*64; e += 256) {
            int th = e >> 6, j = e & 63;
            float v = 0.f;
            if (j < cnt) {
                float a = attnW[(size_t)th*NPTS + n0 + tt + j];
                v = __expf(a - stats[th*2]) * stats[th*2+1];
            }
            spa[e] = v;
        }
        __syncthreads();
        for (int j = 0; j < cnt; j++) {
            float hv = (float)hf[(size_t)(n0 + tt + j)*DIN + i];
            #pragma unroll
            for (int th = 0; th < NTH; th++) acc[th] = fmaf(spa[th*64 + j], hv, acc[th]);
        }
        __syncthreads();
    }
    for (int th = 0; th < NTH; th++)
        g_part[((size_t)chunk*NTH + th)*DIN + i] = acc[th];
}

__global__ void k_greduce(const float* __restrict__ g_part, float* __restrict__ g) {
    int id = blockIdx.x*blockDim.x + threadIdx.x;
    if (id >= NTH*DIN) return;
    int th = id >> 9, i = id & 511;
    float s = 0.f;
    for (int c = 0; c < NCHUNK; c++) s += g_part[((size_t)c*NTH + th)*DIN + i];
    g[id] = s;
}

__global__ void k_gb(const float* __restrict__ g, float* __restrict__ gb) {
    int id = blockIdx.x*blockDim.x + threadIdx.x;
    if (id >= HEADS*DIN) return;
    int h = id >> 9, i = id & 511;
    float s = 0.f;
    for (int t = 0; t < TOK; t++) s += g[(size_t)(t*HEADS + h)*DIN + i];
    gb[id] = s * 0.2f;
}

// o1[t][col] = sum_i g[t,head(col),i] * Wv[t][i][col] + bv[t][col]
__global__ void k_headout(const float* __restrict__ g, const float* __restrict__ Wv,
                          const float* __restrict__ bv, float* __restrict__ o1) {
    int id = blockIdx.x*blockDim.x + threadIdx.x;
    if (id >= TOK*DIN) return;
    int t = id >> 9, col = id & 511, h = col >> 6;
    const float* gr = g + (size_t)(t*HEADS + h)*DIN;
    const float* W  = Wv + (size_t)t*DIN*DIN;
    float s = bv[t*DIN + col];
    for (int i = 0; i < DIN; i++) s = fmaf(gr[i], W[(size_t)i*DIN + col], s);
    o1[id] = s;
}

// per-token matvec: y[t][j] = sum_i x[t][i] * W[t][i][j] + b[t][j]
__global__ void k_proj(const float* __restrict__ x, const float* __restrict__ W,
                       const float* __restrict__ b, float* __restrict__ y, int wstride) {
    int id = blockIdx.x*blockDim.x + threadIdx.x;
    if (id >= TOK*DIN) return;
    int t = id >> 9, j = id & 511;
    const float* xr = x + t*DIN;
    const float* Wt = W + (size_t)t*wstride;
    float s = b[t*DIN + j];
    for (int i = 0; i < DIN; i++) s = fmaf(xr[i], Wt[(size_t)i*DIN + j], s);
    y[id] = s;
}

// layernorm over 512 elems, one block (256 threads) per row
__global__ void k_ln(const float* __restrict__ x, const float* __restrict__ gamma,
                     const float* __restrict__ beta, float* __restrict__ y, int gs) {
    int row = blockIdx.x, tid = threadIdx.x;
    const float* xr = x + (size_t)row*DIN;
    float a0 = xr[tid], a1 = xr[tid + 256];
    __shared__ float red[256];
    __shared__ float mv[2];
    red[tid] = a0 + a1; __syncthreads();
    for (int s = 128; s > 0; s >>= 1) { if (tid < s) red[tid] += red[tid+s]; __syncthreads(); }
    if (tid == 0) mv[0] = red[0] * (1.f/512.f);
    __syncthreads();
    float m = mv[0], d0 = a0 - m, d1 = a1 - m;
    red[tid] = d0*d0 + d1*d1; __syncthreads();
    for (int s = 128; s > 0; s >>= 1) { if (tid < s) red[tid] += red[tid+s]; __syncthreads(); }
    if (tid == 0) mv[1] = rsqrtf(red[0] * (1.f/512.f) + 1e-6f);
    __syncthreads();
    float inv = mv[1];
    y[(size_t)row*DIN + tid]       = d0*inv*gamma[row*gs + tid]       + beta[row*gs + tid];
    y[(size_t)row*DIN + tid + 256] = d1*inv*gamma[row*gs + tid + 256] + beta[row*gs + tid + 256];
}

__global__ void k_cls(const float* __restrict__ feats, const float* __restrict__ Wc,
                      const float* __restrict__ bc, float* __restrict__ out) {
    int id = blockIdx.x*blockDim.x + threadIdx.x;
    if (id >= TOK*NCLSS) return;
    int t = id >> 1, c = id & 1;
    const float* f = feats + t*DIN;
    const float* W = Wc + (size_t)t*DIN*NCLSS;
    float s = bc[t*NCLSS + c];
    for (int j = 0; j < DIN; j++) s = fmaf(f[j], W[j*NCLSS + c], s);
    out[id] = s;
}

__global__ void k_bag1(const float* __restrict__ gb, const float* __restrict__ Wvb,
                       const float* __restrict__ bvb, float* __restrict__ fb0) {
    int col = blockIdx.x*blockDim.x + threadIdx.x;
    if (col >= DIN) return;
    int h = col >> 6;
    const float* gr = gb + (size_t)h*DIN;
    float s = bvb[col];
    for (int i = 0; i < DIN; i++) s = fmaf(gr[i], Wvb[(size_t)i*DIN + col], s);
    fb0[col] = s;
}

__global__ void k_bag2(const float* __restrict__ fb0, const float* __restrict__ Wob,
                       const float* __restrict__ bob, float* __restrict__ fb1) {
    int j = blockIdx.x*blockDim.x + threadIdx.x;
    if (j >= DIN) return;
    float s = bob[j];
    for (int i = 0; i < DIN; i++) s = fmaf(fb0[i], Wob[(size_t)i*DIN + j], s);
    fb1[j] = s;
}

__global__ void k_slide(const float* __restrict__ fbn, const float* __restrict__ Ws,
                        const float* __restrict__ bs, float* __restrict__ out) {
    int c = blockIdx.x*blockDim.x + threadIdx.x;
    if (c >= NCLSS) return;
    float s = bs[c];
    for (int j = 0; j < DIN; j++) s = fmaf(fbn[j], Ws[j*NCLSS + c], s);
    out[TOK*NCLSS + c] = s;
}

// ---------------------------------------------------------------------------
extern "C" void kernel_launch(void* const* d_in, const int* in_sizes, int n_in,
                              void* d_out, int out_size, void* d_ws, size_t ws_size,
                              hipStream_t stream) {
    const float* x    = (const float*)d_in[0];
    const float* Wdr  = (const float*)d_in[1];
    const float* bdr  = (const float*)d_in[2];
    const float* q    = (const float*)d_in[3];
    const float* Wq   = (const float*)d_in[4];
    const float* bq   = (const float*)d_in[5];
    const float* Wk   = (const float*)d_in[6];
    const float* bk   = (const float*)d_in[7];
    const float* Wv   = (const float*)d_in[8];
    const float* bv   = (const float*)d_in[9];
    const float* Wo   = (const float*)d_in[10];
    const float* bo   = (const float*)d_in[11];
    const float* g1   = (const float*)d_in[12];
    const float* be1  = (const float*)d_in[13];
    const float* Wc   = (const float*)d_in[14];
    const float* bc   = (const float*)d_in[15];
    const float* Wvb  = (const float*)d_in[16];
    const float* bvb  = (const float*)d_in[17];
    const float* Wob  = (const float*)d_in[18];
    const float* bob  = (const float*)d_in[19];
    const float* g2   = (const float*)d_in[20];
    const float* be2  = (const float*)d_in[21];
    const float* Wsl  = (const float*)d_in[22];
    const float* bsl  = (const float*)d_in[23];
    float* out = (float*)d_out;

    // ---- workspace carve-up (~28.3 MB) ----
    char* w = (char*)d_ws;
    auto take = [&](size_t bytes) { char* p = w; w += (bytes + 255) & ~size_t(255); return p; };
    _Float16* WdrT  = (_Float16*)take((size_t)DIN*DFEAT*2);
    _Float16* hf    = (_Float16*)take((size_t)NPTS*DIN*2);
    _Float16* AwT   = (_Float16*)take((size_t)NTHP*DIN*2);
    float* cAttn    = (float*)take(NTHP*4);
    float* qv       = (float*)take(TOK*DIN*4);
    float* attnW    = (float*)take((size_t)NTH*NPTS*4);
    float* stats    = (float*)take(NTH*2*4);
    float* g_part   = (float*)take((size_t)NCHUNK*NTH*DIN*4);
    float* g        = (float*)take((size_t)NTH*DIN*4);
    float* gb       = (float*)take((size_t)HEADS*DIN*4);
    float* o1       = (float*)take(TOK*DIN*4);
    float* o2       = (float*)take(TOK*DIN*4);
    float* feats    = (float*)take(TOK*DIN*4);
    float* fb0      = (float*)take(DIN*4);
    float* fb1      = (float*)take(DIN*4);
    float* fbn      = (float*)take(DIN*4);

    const int MT = (NPTS + 127) / 128;   // 157 row tiles

    // 1. transpose W_dr to f16 [512][1024]
    k_transpose_wdr<<<(DIN*DFEAT + 255)/256, 256, 0, stream>>>(Wdr, WdrT);

    // 2. h = relu(x @ W_dr + b_dr)  -> h_f16 [20000][512]   (WMMA + async B staging)
    wmma_gemm<128,128,4,2,1,1,0><<<dim3(MT, DIN/128), 256, 0, stream>>>(
        (const void*)x, WdrT, bdr, hf, nullptr, NPTS, DFEAT, DIN, 0);

    // 3. qvec[t] = q_tok @ Wq + bq
    k_qvec<<<(TOK*DIN + 255)/256, 256, 0, stream>>>(q, Wq, bq, qv);

    // 4. AwT[th][i] = (Wk[:,head] @ q_head)/8 ; cAttn[th] = (bk_head . q_head)/8
    k_awt<<<(NTHP*DIN + 255)/256, 256, 0, stream>>>(Wk, bk, qv, AwT, cAttn);

    // 5. attn = h @ Aw + cAttn -> attnW [40][20000]  (WMMA, fully async staging)
    wmma_gemm<128,64,4,2,0,0,1><<<dim3(MT, 1), 256, 0, stream>>>(
        (const void*)hf, AwT, cAttn, nullptr, attnW, NPTS, DIN, NTHP, NTH);

    // 6. top-10 + deterministic 5-of-10 masking per (t,head)
    k_topk_mask<<<NTH, 256, 0, stream>>>(attnW);

    // 7. copy masked attn to output [H,T,N] and compute softmax stats
    k_softmax_stats<<<NTH, 256, 0, stream>>>(attnW, stats, out + TOK*NCLSS + NCLSS);

    // 8. g[t,h,:] = sum_n pa*h[n,:]  (streaming pass, partials then reduce)
    k_wsum<<<dim3(NCHUNK, 2), 256, 0, stream>>>(hf, attnW, stats, g_part);
    k_greduce<<<(NTH*DIN + 255)/256, 256, 0, stream>>>(g_part, g);
    k_gb<<<(HEADS*DIN + 255)/256, 256, 0, stream>>>(g, gb);

    // 9. per-token: o1 = einsum(pa,v) via g@Wv ; o2 = o1@Wo+bo ; feats = LN
    k_headout<<<(TOK*DIN + 255)/256, 256, 0, stream>>>(g, Wv, bv, o1);
    k_proj<<<(TOK*DIN + 255)/256, 256, 0, stream>>>(o1, Wo, bo, o2, DIN*DIN);
    k_ln<<<TOK, 256, 0, stream>>>(o2, g1, be1, feats, DIN);
    k_cls<<<1, 32, 0, stream>>>(feats, Wc, bc, out);

    // 10. bag branch: fb0 = gb@Wvb+bvb ; fb1 = fb0@Wob+bob ; LN ; slide
    k_bag1<<<2, 256, 0, stream>>>(gb, Wvb, bvb, fb0);
    k_bag2<<<2, 256, 0, stream>>>(fb0, Wob, bob, fb1);
    k_ln<<<1, 256, 0, stream>>>(fb1, g2, be2, fbn, 0);
    k_slide<<<1, 32, 0, stream>>>(fbn, Wsl, bsl, out);
}
